// Model41_30021821399564
// MI455X (gfx1250) — matrix operand order repaired and checked by reading, not compiled, not run on hardware
//
#include <hip/hip_runtime.h>
#include <hip/hip_bf16.h>
#include <math.h>

typedef __bf16 bf16_t;
typedef __attribute__((ext_vector_type(16))) __bf16 v16bf;
typedef __attribute__((ext_vector_type(8)))  __bf16 v8bf;
typedef __attribute__((ext_vector_type(8)))  float  v8f;

#define B_SZ   128
#define T_SZ   512
#define V_SZ   50000
#define E_SZ   300
#define H_SZ   256
#define OUT_SZ 5
#define FOURH  1024
#define EPAD   320                      // E padded to multiple of 32
#define ROWTILES (T_SZ * B_SZ / 16)     // 4096
#define COLTILES (FOURH / 16)           // 64
#define CG_SIZE  8                      // col tiles per wave (shared A fragment)
#define NCG      (COLTILES / CG_SIZE)   // 8 col groups
#define TILES_PER_DIR (ROWTILES * COLTILES) // 262144

union V16U { v16bf v; v8bf h[2]; };

__device__ __forceinline__ float sigmoidf_(float x) { return 1.0f / (1.0f + __expf(-x)); }

// ---------------------------------------------------------------------------
// Kernel 0: one-time conversions to bf16:
//   emb  [V,300] fp32 -> emb_bf [V,320] bf16 (zero-padded K)
//   W_ih [4H,300] -> [4H,320] bf16 (both dirs), W_hh [4H,256] bf16 (both dirs)
// ---------------------------------------------------------------------------
__global__ void k_prep(const float* __restrict__ emb,
                       const float* __restrict__ wih_f, const float* __restrict__ wih_b,
                       const float* __restrict__ whh_f, const float* __restrict__ whh_b,
                       bf16_t* __restrict__ emb_bf,
                       bf16_t* __restrict__ wih_bf, bf16_t* __restrict__ whh_bf) {
  int idx = blockIdx.x * blockDim.x + threadIdx.x;
  const int nemb = V_SZ * EPAD;
  const int nwih = 2 * FOURH * EPAD;
  const int nwhh = 2 * FOURH * H_SZ;
  if (idx < nemb) {
    int row = idx / EPAD, k = idx % EPAD;
    emb_bf[idx] = (bf16_t)((k < E_SZ) ? emb[row * E_SZ + k] : 0.0f);
    return;
  }
  idx -= nemb;
  if (idx < nwih) {
    int dir = idx / (FOURH * EPAD);
    int r   = idx % (FOURH * EPAD);
    int n = r / EPAD, k = r % EPAD;
    const float* src = dir ? wih_b : wih_f;
    wih_bf[idx] = (bf16_t)((k < E_SZ) ? src[n * E_SZ + k] : 0.0f);
    return;
  }
  idx -= nwih;
  if (idx < nwhh) {
    int dir = idx / (FOURH * H_SZ);
    int r   = idx % (FOURH * H_SZ);
    const float* src = dir ? whh_b : whh_f;
    whh_bf[idx] = (bf16_t)src[r];
  }
}

// ---------------------------------------------------------------------------
// Kernel 1: xg[dir][row=t*B+b][4H] = emb_bf[x[b,t']] @ W_ih^T + bias (bf16 WMMA)
// Each wave: one 16-row A fragment reused across CG_SIZE=8 output tiles.
// All 8 B fragments of a K-chunk are staged before the WMMA burst so the
// scheduler emits one load clause draining into 8 independent v_wmma issues.
// dir==1 gathers time-reversed rows (packed-sequence semantics).
// Output stored fragment-native: tile = 512B = 32 lanes x 8 bf16.
// ---------------------------------------------------------------------------
__global__ void __launch_bounds__(256) k_inproj(
    const int* __restrict__ x, const int* __restrict__ lengths,
    const bf16_t* __restrict__ emb_bf, const bf16_t* __restrict__ wih_bf,
    const float* __restrict__ bias_f, const float* __restrict__ bias_b,
    bf16_t* __restrict__ xg) {
  const int dir    = blockIdx.y;
  const int lane   = threadIdx.x & 31;
  const int waveId = threadIdx.x >> 5;
  const int flat   = blockIdx.x * 8 + waveId;   // [0, 32768)
  const int cgrp   = flat & (NCG - 1);
  const int rowTile = flat / NCG;
  const int mr   = lane & 15;
  const int nsub = lane & 15;
  const int kb   = (lane < 16) ? 0 : 8;    // A-frag K sub-offset (ISA 7.12.2)
  const int bkb  = (lane < 16) ? 0 : 8;    // B-frag K sub-offset

  // A-row for this lane: row = t*B + b
  int row = rowTile * 16 + mr;
  int t = row >> 7;          // / 128
  int b = row & 127;
  int tt = t;
  if (dir) {                 // rev = clip(len-1-t, 0, T-1)
    int len = lengths[b];
    tt = len - 1 - t;
    tt = tt < 0 ? 0 : (tt > T_SZ - 1 ? T_SZ - 1 : tt);
  }
  const int tok = x[b * T_SZ + tt];
  const bf16_t* erow  = emb_bf + (size_t)tok * EPAD;
  const bf16_t* wbase = wih_bf + ((size_t)dir * FOURH + cgrp * (CG_SIZE * 16) + nsub) * EPAD;

  v8f acc[CG_SIZE];
#pragma unroll
  for (int cg = 0; cg < CG_SIZE; ++cg)
#pragma unroll
    for (int i = 0; i < 8; ++i) acc[cg][i] = 0.0f;

  for (int kc = 0; kc < EPAD / 32; ++kc) {
    const int k0  = kc * 32 + kb;
    const int bko = kc * 32 + bkb;
    V16U a;                                  // shared A fragment (bf16, no cvt)
    a.h[0] = *(const v8bf*)(erow + k0);
    a.h[1] = *(const v8bf*)(erow + k0 + 16);
    V16U bbs[CG_SIZE];                       // stage all B fragments first
#pragma unroll
    for (int cg = 0; cg < CG_SIZE; ++cg) {
      const bf16_t* bp = wbase + (size_t)cg * 16 * EPAD + bko;
      bbs[cg].h[0] = *(const v8bf*)bp;
      bbs[cg].h[1] = *(const v8bf*)(bp + 16);
    }
#pragma unroll
    for (int cg = 0; cg < CG_SIZE; ++cg)     // 8 independent WMMAs back-to-back
      acc[cg] = __builtin_amdgcn_wmma_f32_16x16x32_bf16(false, a.v, false, bbs[cg].v,
                                                        (short)0, acc[cg], false, false);
  }

  const float* bias = dir ? bias_b : bias_f;
#pragma unroll
  for (int cg = 0; cg < CG_SIZE; ++cg) {
    const int colTile = cgrp * CG_SIZE + cg;
    const float bv = bias[colTile * 16 + nsub];
    v8bf outv;
#pragma unroll
    for (int i = 0; i < 8; ++i) outv[i] = (bf16_t)(acc[cg][i] + bv);
    bf16_t* dst = xg + ((size_t)dir * TILES_PER_DIR + (size_t)rowTile * COLTILES + colTile) * 256
                  + lane * 8;
    *(v8bf*)dst = outv;
  }
}

// ---------------------------------------------------------------------------
// Kernel 2: persistent LSTM recurrence. grid = (8 batch-blocks, 2 dirs),
// block = 512 thr = 16 waves. Wave w owns hidden columns [w*16, w*16+16) and
// its 4 gate tiles; h double-buffered in LDS (bf16), c in VGPRs (f32).
// All pooling (first/second/mean/max with reference quirks) fused in.
// ---------------------------------------------------------------------------
__global__ void __launch_bounds__(512, 1) k_recur(
    const int* __restrict__ lengths, const bf16_t* __restrict__ whh_bf,
    const bf16_t* __restrict__ xg, float* __restrict__ rep) {
  const int bTile = blockIdx.x;       // 0..7
  const int dir   = blockIdx.y;       // 0..1
  const int lane  = threadIdx.x & 31;
  const int jTile = threadIdx.x >> 5; // 0..15
  const int mr   = lane & 15;
  const int hi8  = (lane >> 4) & 1;
  const int kb   = (lane < 16) ? 0 : 8;
  const int bkb  = (lane < 16) ? 0 : 8;
  const int nsub = lane & 15;

  __shared__ __align__(16) bf16_t hbuf[2][16][H_SZ];
  for (int i = threadIdx.x; i < 2 * 16 * H_SZ; i += 512)
    ((bf16_t*)hbuf)[i] = (bf16_t)0.0f;
  __syncthreads();

  float cst[8], sum[8], mx[8], fst[8];
  int lenv[8], trig[8];
#pragma unroll
  for (int v = 0; v < 8; ++v) {
    cst[v] = 0.0f; sum[v] = 0.0f; mx[v] = -3.4e38f; fst[v] = 0.0f;
    const int b  = bTile * 16 + v + hi8 * 8;
    const int lb = lengths[b];
    lenv[v] = lb;
    if (dir == 0) {
      trig[v] = lb - 1;                       // first = hf[b, len_b-1]
    } else {                                  // second[B-1-b] = hb[b, len_{B-1-b}-1]
      const int lp = lengths[B_SZ - 1 - b];   //   valid iff lp <= lb, at scan step lb-lp
      trig[v] = (lp <= lb) ? (lb - lp) : -1;
    }
  }

  const bf16_t* whh = whh_bf + (size_t)dir * FOURH * H_SZ;
  const bf16_t* xgd = xg + (size_t)dir * (size_t)TILES_PER_DIR * 256;

  for (int s = 0; s < T_SZ; ++s) {
    const int cur = s & 1, nxt = cur ^ 1;
    const size_t rowTile = (size_t)(s * 8 + bTile);

    v8f acc[4];
#pragma unroll
    for (int g = 0; g < 4; ++g) {            // C init = precomputed xg fragment
      const size_t tile = rowTile * COLTILES + (g * 16 + jTile);
      const v8bf xf = *(const v8bf*)(xgd + tile * 256 + lane * 8);
#pragma unroll
      for (int i = 0; i < 8; ++i) acc[g][i] = (float)xf[i];
    }
    if (s + 1 < T_SZ) {                      // prefetch next step's xg tiles
      const size_t rtn = (size_t)((s + 1) * 8 + bTile);
#pragma unroll
      for (int g = 0; g < 4; ++g)
        __builtin_prefetch(xgd + (rtn * COLTILES + (g * 16 + jTile)) * 256 + lane * 8, 0, 1);
    }

#pragma unroll
    for (int kc = 0; kc < H_SZ / 32; ++kc) { // h @ W_hh^T : 8 chunks x 4 gates
      V16U a;
      const int k0 = kc * 32 + kb;
      a.h[0] = *(const v8bf*)&hbuf[cur][mr][k0];
      a.h[1] = *(const v8bf*)&hbuf[cur][mr][k0 + 16];
      V16U bbs[4];                           // stage 4 gate B fragments first
#pragma unroll
      for (int g = 0; g < 4; ++g) {
        const int n = g * H_SZ + jTile * 16 + nsub;
        const bf16_t* bp = whh + (size_t)n * H_SZ + kc * 32 + bkb;
        bbs[g].h[0] = *(const v8bf*)bp;
        bbs[g].h[1] = *(const v8bf*)(bp + 16);
      }
#pragma unroll
      for (int g = 0; g < 4; ++g)            // 4 independent WMMAs back-to-back
        acc[g] = __builtin_amdgcn_wmma_f32_16x16x32_bf16(false, a.v, false, bbs[g].v,
                                                         (short)0, acc[g], false, false);
    }

#pragma unroll
    for (int v = 0; v < 8; ++v) {            // gate nonlinearity + state + pooling
      const float gi = sigmoidf_(acc[0][v]);
      const float gf = sigmoidf_(acc[1][v]);
      const float gg = tanhf(acc[2][v]);
      const float go = sigmoidf_(acc[3][v]);
      const float c  = gf * cst[v] + gi * gg;
      cst[v] = c;
      const float h = go * tanhf(c);
      if (s < lenv[v]) { sum[v] += h; mx[v] = fmaxf(mx[v], h); }
      if (s == trig[v]) fst[v] = h;
      hbuf[nxt][v + hi8 * 8][jTile * 16 + nsub] = (bf16_t)h;
    }
    __syncthreads();
  }

#pragma unroll
  for (int v = 0; v < 8; ++v) {
    const int b = bTile * 16 + v + hi8 * 8;
    const int j = jTile * 16 + nsub;
    const float mxv = (lenv[v] < T_SZ) ? fmaxf(mx[v], 0.0f) : mx[v]; // max incl. pad zeros
    if (dir == 0) {
      rep[(size_t)b * 1536 + j]       = fst[v];
      rep[(size_t)b * 1536 + 256 + j] = sum[v] / (float)lenv[v];
      rep[(size_t)b * 1536 + 512 + j] = mxv;
    } else {
      rep[(size_t)(B_SZ - 1 - b) * 1536 + 768 + j] = fst[v];        // batch-reversed quirk
      rep[(size_t)b * 1536 + 1024 + j] = sum[v] / (float)lengths[B_SZ - 1 - b];
      rep[(size_t)b * 1536 + 1280 + j] = mxv;
    }
  }
}

// ---------------------------------------------------------------------------
// Kernel 3: out[b,o] = rep[b,:] . lin_w[o,:] + lin_b[o]   (640 dots of K=1536)
// ---------------------------------------------------------------------------
__global__ void k_out(const float* __restrict__ rep, const float* __restrict__ lin_w,
                      const float* __restrict__ lin_b, float* __restrict__ out) {
  const int idx = blockIdx.x * blockDim.x + threadIdx.x;
  if (idx >= B_SZ * OUT_SZ) return;
  const int b = idx / OUT_SZ, o = idx % OUT_SZ;
  const float* r = rep + (size_t)b * 1536;
  const float* w = lin_w + (size_t)o * 1536;
  float acc = lin_b[o];
  for (int k = 0; k < 1536; ++k) acc += r[k] * w[k];
  out[idx] = acc;
}

// ---------------------------------------------------------------------------
extern "C" void kernel_launch(void* const* d_in, const int* in_sizes, int n_in,
                              void* d_out, int out_size, void* d_ws, size_t ws_size,
                              hipStream_t stream) {
  const int*   x       = (const int*)d_in[0];
  const int*   lengths = (const int*)d_in[1];
  const float* emb     = (const float*)d_in[2];
  const float* w_ih_f  = (const float*)d_in[3];
  const float* w_hh_f  = (const float*)d_in[4];
  const float* b_f     = (const float*)d_in[5];
  const float* w_ih_b  = (const float*)d_in[6];
  const float* w_hh_b  = (const float*)d_in[7];
  const float* b_b     = (const float*)d_in[8];
  const float* lin_w   = (const float*)d_in[9];
  const float* lin_b   = (const float*)d_in[10];
  float* out = (float*)d_out;

  char* ws = (char*)d_ws;
  bf16_t* emb_bf = (bf16_t*)(ws);                 // 50000*320*2 = 32,000,000 B
  bf16_t* wih_bf = (bf16_t*)(ws + 32000000);      // 2*1024*320*2 = 1,310,720 B
  bf16_t* whh_bf = (bf16_t*)(ws + 33310720);      // 2*1024*256*2 = 1,048,576 B
  float*  rep    = (float*) (ws + 34359296);      // 128*1536*4   =   786,432 B
  bf16_t* xg     = (bf16_t*)(ws + 35145728);      // 2*262144*512 = 268,435,456 B

  const int prepN = V_SZ * EPAD + 2 * FOURH * EPAD + 2 * FOURH * H_SZ;
  k_prep<<<dim3((prepN + 255) / 256), dim3(256), 0, stream>>>(
      emb, w_ih_f, w_ih_b, w_hh_f, w_hh_b, emb_bf, wih_bf, whh_bf);

  k_inproj<<<dim3(ROWTILES * NCG / 8, 2), dim3(256), 0, stream>>>(
      x, lengths, emb_bf, wih_bf, b_f, b_b, xg);

  k_recur<<<dim3(8, 2), dim3(512), 0, stream>>>(lengths, whh_bf, xg, rep);

  k_out<<<dim3((B_SZ * OUT_SZ + 127) / 128), dim3(128), 0, stream>>>(
      rep, lin_w, lin_b, out);
}
